// AttentionModule_72060961292289
// MI455X (gfx1250) — matrix operand order repaired and checked by reading, not compiled, not run on hardware
//
#include <hip/hip_runtime.h>
#include <hip/hip_bf16.h>

#define DIM 4096
#define SEQ 2048
#define NH 32
#define NKV 8
#define HD 128
#define ROPE_THETA 50000.0f

typedef __attribute__((ext_vector_type(16))) _Float16 v16h;
typedef __attribute__((ext_vector_type(8)))  _Float16 v8h;
typedef __attribute__((ext_vector_type(8)))  float    v8f;

// ---------------------------------------------------------------------------
// WMMA fragment helpers (contiguous k order, same in A and B, wave32)
// ---------------------------------------------------------------------------

// A-matrix 16x32 f16 (MxK): lane L holds row M=L&15; if L<16 K range
// 0..15, else 16..31 (matches load_b_f16); packed 2 f16 per VGPR.
__device__ inline v16h load_a_f16(const _Float16* base, int ld, int row0, int k0) {
  int lane = threadIdx.x & 31;
  int m  = lane & 15;
  int ks = (lane >> 4) * 16;
  const _Float16* p = base + (size_t)(row0 + m) * ld + k0 + ks;
  v8h lo = *(const v8h*)p;
  v8h hi = *(const v8h*)(p + 8);
  v16h a;
#pragma unroll
  for (int i = 0; i < 8; ++i) { a[i] = lo[i]; a[i + 8] = hi[i]; }
  return a;
}

// B-matrix 32x16 f16 (KxN) where B[k][n] = W[n0+n][k0+k] (W row-major NxK):
// lane L holds column n=L&15, K range (L>=16 ? 16..31 : 0..15) -> one
// contiguous 32-byte load per lane.
__device__ inline v16h load_b_f16(const _Float16* base, int ld, int n0, int k0) {
  int lane = threadIdx.x & 31;
  int n  = lane & 15;
  int kb = (lane >> 4) * 16;
  return *(const v16h*)(base + (size_t)(n0 + n) * ld + k0 + kb);
}

__device__ inline v8f wmma_f16(v16h a, v16h b, v8f c) {
  return __builtin_amdgcn_wmma_f32_16x16x32_f16(false, a, false, b,
                                                (short)0, c, false, false);
}

// ---------------------------------------------------------------------------
// RMSNorm -> fp16
// ---------------------------------------------------------------------------
__global__ void __launch_bounds__(256)
rmsnorm_kernel(const float* __restrict__ hidden, const float* __restrict__ norm_w,
               _Float16* __restrict__ h_out) {
  int row = blockIdx.x;
  const float* x = hidden + (size_t)row * DIM;
  __shared__ float red[256];
  float ss = 0.f;
  for (int i = threadIdx.x; i < DIM; i += 256) { float v = x[i]; ss += v * v; }
  red[threadIdx.x] = ss;
  __syncthreads();
  for (int s = 128; s > 0; s >>= 1) {
    if (threadIdx.x < (unsigned)s) red[threadIdx.x] += red[threadIdx.x + s];
    __syncthreads();
  }
  float inv = rsqrtf(red[0] / (float)DIM + 1e-5f);
  _Float16* o = h_out + (size_t)row * DIM;
  for (int i = threadIdx.x; i < DIM; i += 256)
    o[i] = (_Float16)(x[i] * inv * norm_w[i]);
}

// ---------------------------------------------------------------------------
// fp32 -> fp16 conversion (weights)
// ---------------------------------------------------------------------------
__global__ void __launch_bounds__(256)
f32_to_f16(const float* __restrict__ in, _Float16* __restrict__ out, size_t n) {
  size_t i = (size_t)blockIdx.x * 256u + threadIdx.x;
  size_t stride = (size_t)gridDim.x * 256u;
  for (; i < n; i += stride) out[i] = (_Float16)in[i];
}

// ---------------------------------------------------------------------------
// GEMM: C[M,N] = A[M,K] (f16, row-major) x W[N,K]^T (f16, row-major), f32 out.
// One wave per block; 64x64 tile per wave: 16 WMMAs per K-step against
// 8 fragment loads (2x the reuse of a 32x64 tile; L2-bandwidth bound).
// ---------------------------------------------------------------------------
__global__ void __launch_bounds__(32)
gemm_f16f32(const _Float16* __restrict__ A, const _Float16* __restrict__ W,
            float* __restrict__ C, int M, int N, int K) {
  int col0 = blockIdx.x * 64;
  int row0 = blockIdx.y * 64;
  v8f acc[4][4] = {};
  for (int k0 = 0; k0 < K; k0 += 32) {
    v16h a[4], b[4];
#pragma unroll
    for (int i = 0; i < 4; ++i) a[i] = load_a_f16(A, K, row0 + 16 * i, k0);
#pragma unroll
    for (int j = 0; j < 4; ++j) b[j] = load_b_f16(W, K, col0 + 16 * j, k0);
#pragma unroll
    for (int i = 0; i < 4; ++i)
#pragma unroll
      for (int j = 0; j < 4; ++j)
        acc[i][j] = wmma_f16(a[i], b[j], acc[i][j]);
  }
  int lane = threadIdx.x & 31;
  int n  = lane & 15;
  int mb = (lane >> 4) * 8;
#pragma unroll
  for (int i = 0; i < 4; ++i)
#pragma unroll
    for (int j = 0; j < 4; ++j)
#pragma unroll
      for (int r = 0; r < 8; ++r)
        C[(size_t)(row0 + 16 * i + mb + r) * N + col0 + 16 * j + n] = acc[i][j][r];
}

// ---------------------------------------------------------------------------
// RoPE: x [SEQ][n_heads*HD] f32 -> out [n_heads][SEQ][HD] f16 (head-major)
// ---------------------------------------------------------------------------
__global__ void __launch_bounds__(64)
rope_kernel(const float* __restrict__ x, _Float16* __restrict__ out, int n_heads) {
  int s = blockIdx.x, h = blockIdx.y, d2 = threadIdx.x; // d2: 0..63
  const float* p = x + ((size_t)s * n_heads + h) * HD;
  float xr = p[2 * d2], xi = p[2 * d2 + 1];
  float inv_freq = powf(ROPE_THETA, -(2.0f * d2) / (float)HD);
  float ang = (float)s * inv_freq;
  float c = cosf(ang), sn = sinf(ang);
  _Float16* o = out + ((size_t)h * SEQ + s) * HD;
  o[2 * d2]     = (_Float16)(xr * c - xi * sn);
  o[2 * d2 + 1] = (_Float16)(xr * sn + xi * c);
}

// V: xv [SEQ][NKV*HD] f32 -> vt [NKV][HD][SEQ] f16 (d-major for P*V B-frags)
__global__ void __launch_bounds__(128)
v_transpose(const float* __restrict__ xv, _Float16* __restrict__ vt) {
  int s = blockIdx.x, h = blockIdx.y, d = threadIdx.x;
  vt[((size_t)h * HD + d) * SEQ + s] =
      (_Float16)xv[((size_t)s * NKV + h) * HD + d];
}

// ---------------------------------------------------------------------------
// Causal attention, one (head, 16-row q-tile) per 4-wave workgroup.
// Score panel stored as f16 in LDS (66KB -> 4 workgroups/WGP on 320KB LDS);
// P*V A-fragments read straight from LDS via ds_load_b128.
// ---------------------------------------------------------------------------
#define SSTRIDE (SEQ + 32)   // f16 elements; row = 4160B, keeps 16B alignment
__global__ void __launch_bounds__(128)
attn_kernel(const _Float16* __restrict__ qf, const _Float16* __restrict__ kf,
            const _Float16* __restrict__ vt, _Float16* __restrict__ attn_out) {
  int qt = blockIdx.x;           // q tile 0..127
  int h  = blockIdx.y;           // head 0..31
  int kvh = h >> 2;              // GQA: 4 q-heads per kv-head
  int q0 = qt * 16;
  int tid  = threadIdx.x;
  int wave = tid >> 5;
  int lane = tid & 31;

  __shared__ _Float16 S16[16][SSTRIDE];
  __shared__ float redbuf[16][8];
  __shared__ float rowmax[16];
  __shared__ float rowsum[16];

  const _Float16* Q  = qf + (size_t)h   * SEQ * HD;
  const _Float16* Kh = kf + (size_t)kvh * SEQ * HD;
  const _Float16* Vh = vt + (size_t)kvh * HD * SEQ;

  // Q fragments for this tile: 4 chunks of head_dim (contraction K=128)
  v16h aq[4];
#pragma unroll
  for (int dk = 0; dk < 4; ++dk) aq[dk] = load_a_f16(Q, HD, q0, dk * 32);

  const int njt = qt + 1;                   // causal: k tiles 0..qt
  const float scale = 0.08838834764831845f; // 1/sqrt(128)

  // ---- pass 1: S = scale * Q*K^T, causal mask, f16 into LDS ----
  for (int j = wave; j < njt; j += 4) {
    v8f acc = {};
#pragma unroll
    for (int dk = 0; dk < 4; ++dk) {
      v16h b = load_b_f16(Kh, HD, j * 16, dk * 32);
      acc = wmma_f16(aq[dk], b, acc);
    }
    int n  = lane & 15;
    int mb = (lane >> 4) * 8;
#pragma unroll
    for (int r = 0; r < 8; ++r) {
      int row = mb + r;
      float v = acc[r] * scale;
      if (j == qt && (16 * j + n) > (q0 + row)) v = -30000.0f; // f16-safe -inf
      S16[row][16 * j + n] = (_Float16)v;
    }
  }
  __syncthreads();

  // ---- pass 2a: row softmax in f32 math over the f16 panel ----
  int ncols = njt * 16;
  int row = tid & 15, sub = tid >> 4;     // 8 lanes per row
  {
    float mx = -30000.0f;
    for (int c = sub; c < ncols; c += 8) mx = fmaxf(mx, (float)S16[row][c]);
    redbuf[row][sub] = mx;
    __syncthreads();
    if (sub == 0) {
      float m = redbuf[row][0];
#pragma unroll
      for (int i = 1; i < 8; ++i) m = fmaxf(m, redbuf[row][i]);
      rowmax[row] = m;
    }
    __syncthreads();
    float m = rowmax[row];
    float sum = 0.f;
    for (int c = sub; c < ncols; c += 8) {
      float p = __expf((float)S16[row][c] - m);
      S16[row][c] = (_Float16)p;          // p in [0,1]: f16-safe
      sum += p;
    }
    redbuf[row][sub] = sum;
    __syncthreads();
    if (sub == 0) {
      float t = 0.f;
#pragma unroll
      for (int i = 0; i < 8; ++i) t += redbuf[row][i];
      rowsum[row] = t;
    }
    // zero the pad half-tile so the K=32 WMMA loop reads zeros there
    int nc32 = (ncols + 31) & ~31;
    for (int c = ncols + sub; c < nc32; c += 8) S16[row][c] = (_Float16)0.f;
  }
  __syncthreads();

  // ---- pass 2b: O = P*V, each wave handles 2 of 8 d-tiles ----
  int nc32 = (ncols + 31) & ~31;
#pragma unroll
  for (int t = 0; t < 2; ++t) {
    int d0 = (wave + 4 * t) * 16;
    v8f acc = {};
    for (int k0 = 0; k0 < nc32; k0 += 32) {
      v16h a = load_a_f16(&S16[0][0], SSTRIDE, 0, k0); // ds_load_b128 x2
      v16h b = load_b_f16(Vh, SEQ, d0, k0); // B[k][n] = V_t[d0+n][k], contiguous
      acc = wmma_f16(a, b, acc);
    }
    int n  = lane & 15;
    int mb = (lane >> 4) * 8;
#pragma unroll
    for (int r = 0; r < 8; ++r) {
      int orow = mb + r;
      attn_out[(size_t)(q0 + orow) * DIM + h * HD + d0 + n] =
          (_Float16)(acc[r] / rowsum[orow]);
    }
  }
}

// ---------------------------------------------------------------------------
// Launch
// ---------------------------------------------------------------------------
extern "C" void kernel_launch(void* const* d_in, const int* in_sizes, int n_in,
                              void* d_out, int out_size, void* d_ws, size_t ws_size,
                              hipStream_t stream) {
  const float* hidden = (const float*)d_in[0];
  const float* norm_w = (const float*)d_in[1];
  const float* wq = (const float*)d_in[2];
  const float* wk = (const float*)d_in[3];
  const float* wv = (const float*)d_in[4];
  const float* wo = (const float*)d_in[5];
  float* out = (float*)d_out;

  char* ws = (char*)d_ws;
  size_t off = 0;
  _Float16* h16   = (_Float16*)(ws + off); off += (size_t)SEQ * DIM * 2;        // 16MB
  _Float16* wq16  = (_Float16*)(ws + off); off += (size_t)DIM * DIM * 2;        // 32MB
  _Float16* wk16  = (_Float16*)(ws + off); off += (size_t)NKV * HD * DIM * 2;   // 8MB
  _Float16* wv16  = (_Float16*)(ws + off); off += (size_t)NKV * HD * DIM * 2;   // 8MB
  _Float16* wo16  = (_Float16*)(ws + off); off += (size_t)DIM * DIM * 2;        // 32MB
  float*    xq    = (float*)   (ws + off); off += (size_t)SEQ * DIM * 4;        // 32MB
  float*    xk    = (float*)   (ws + off); off += (size_t)SEQ * NKV * HD * 4;   // 8MB
  float*    xv    = (float*)   (ws + off); off += (size_t)SEQ * NKV * HD * 4;   // 8MB
  _Float16* q16   = (_Float16*)(ws + off); off += (size_t)NH  * SEQ * HD * 2;   // 16MB
  _Float16* k16   = (_Float16*)(ws + off); off += (size_t)NKV * SEQ * HD * 2;   // 4MB
  _Float16* vt16  = (_Float16*)(ws + off); off += (size_t)NKV * HD * SEQ * 2;   // 4MB
  _Float16* at16  = (_Float16*)(ws + off); off += (size_t)SEQ * DIM * 2;        // 16MB

  // 1) RMSNorm -> fp16 activations
  rmsnorm_kernel<<<SEQ, 256, 0, stream>>>(hidden, norm_w, h16);

  // 2) weights -> fp16
  f32_to_f16<<<4096, 256, 0, stream>>>(wq, wq16, (size_t)DIM * DIM);
  f32_to_f16<<<1024, 256, 0, stream>>>(wk, wk16, (size_t)NKV * HD * DIM);
  f32_to_f16<<<1024, 256, 0, stream>>>(wv, wv16, (size_t)NKV * HD * DIM);
  f32_to_f16<<<4096, 256, 0, stream>>>(wo, wo16, (size_t)DIM * DIM);

  // 3) projections (WMMA GEMMs): xq = h @ wq^T, etc.
  gemm_f16f32<<<dim3(DIM / 64, SEQ / 64), 32, 0, stream>>>(h16, wq16, xq, SEQ, DIM, DIM);
  gemm_f16f32<<<dim3(NKV * HD / 64, SEQ / 64), 32, 0, stream>>>(h16, wk16, xk, SEQ, NKV * HD, DIM);
  gemm_f16f32<<<dim3(NKV * HD / 64, SEQ / 64), 32, 0, stream>>>(h16, wv16, xv, SEQ, NKV * HD, DIM);

  // 4) RoPE + layout changes
  rope_kernel<<<dim3(SEQ, NH),  64, 0, stream>>>(xq, q16, NH);
  rope_kernel<<<dim3(SEQ, NKV), 64, 0, stream>>>(xk, k16, NKV);
  v_transpose<<<dim3(SEQ, NKV), 128, 0, stream>>>(xv, vt16);

  // 5) causal attention (WMMA scores + WMMA P*V)
  attn_kernel<<<dim3(SEQ / 16, NH), 128, 0, stream>>>(q16, k16, vt16, at16);

  // 6) output projection: out = attn @ wo^T (f32 result)
  gemm_f16f32<<<dim3(DIM / 64, SEQ / 64), 32, 0, stream>>>(at16, wo16, out, SEQ, DIM, DIM);
}
